// HyperdimensionalNeuralLayer_50964081934879
// MI455X (gfx1250) — matrix operand order repaired and checked by reading, not compiled, not run on hardware
//
#include <hip/hip_runtime.h>
#include <hip/hip_bf16.h>
#include <math.h>

#define BATCH   8192
#define IN_DIM  2048
#define OUT_DIM 2048
#define DEPTH   5

typedef __attribute__((ext_vector_type(16))) __bf16 v16bf;
typedef __attribute__((ext_vector_type(8)))  __bf16 v8bf;
typedef __attribute__((ext_vector_type(8)))  float  v8f;

// ---------------------------------------------------------------------------
// Kernel 1: hyperdimensional probabilities, reduced to a real 11-wide window
// recurrence per output row (phases are a global per-row phase => dead code
// under |.|^2; roll +/-1 over DEPTH=5 steps => only columns -5..+5 matter,
// and s0 is a uniform constant).
// ---------------------------------------------------------------------------
__global__ void hd_probs_kernel(const float* __restrict__ hdw,
                                float* __restrict__ hd) {
    int m = blockIdx.x * blockDim.x + threadIdx.x;
    if (m >= OUT_DIM) return;

    const float init = 0.02209708691207961f;  // 1/sqrt(2048)
    float w[11];
#pragma unroll
    for (int j = 0; j < 11; ++j) w[j] = init;

#pragma unroll
    for (int d = 0; d < DEPTH; ++d) {
        // angles[d, m] = hd_w[d, m, 0]
        float ang = hdw[(size_t)d * IN_DIM * OUT_DIM + (size_t)m * OUT_DIM];
        float s, c;
        sincosf(ang, &s, &c);
        float nw[11];
#pragma unroll
        for (int j = 1; j <= 9; ++j)
            nw[j] = c * w[j] - s * w[j + 1] + s * w[j - 1];
#pragma unroll
        for (int j = 1; j <= 9; ++j) w[j] = nw[j];
    }
    hd[m] = w[5] * w[5];  // |s[m,0]|^2
}

// ---------------------------------------------------------------------------
// Kernel 2: fused GEMM  out = tanh(x @ W + bias + hd)
// Split-bf16 (hi/lo) WMMA: A*B ~= ah*bh + ah*bl + al*bh  (near-fp32 accuracy)
// Block tile 128x128, BK=32, 8 waves (4x2), each wave -> 2x4 WMMA tiles.
// ---------------------------------------------------------------------------
#define BM 128
#define BN 128
#define BK 32
#define APITCH 40   // bf16 elements per row: 80B rows -> 16B-aligned frags
#define BPITCH 40

__global__ __launch_bounds__(256)
void fused_gemm_tanh_kernel(const float* __restrict__ X,
                            const float* __restrict__ W,
                            const float* __restrict__ bias,
                            const float* __restrict__ hd,
                            float* __restrict__ out) {
    __shared__ __bf16 Ah[BM * APITCH];
    __shared__ __bf16 Al[BM * APITCH];
    __shared__ __bf16 Bh[BN * BPITCH];  // stored transposed: [n][k]
    __shared__ __bf16 Bl[BN * BPITCH];

    const int tid  = threadIdx.x;
    const int lane = tid & 31;
    const int wave = tid >> 5;     // 0..7
    const int wm   = wave >> 1;    // 0..3 (row group of 32)
    const int wn   = wave & 1;     // 0..1 (col group of 64)
    const int half = lane >> 4;    // 0/1
    const int l16  = lane & 15;

    const int bm = blockIdx.y * BM;
    const int bn = blockIdx.x * BN;

    v8f acc[2][4] = {};

    for (int k0 = 0; k0 < IN_DIM; k0 += BK) {
        // ---- stage A tile: 128x32 f32 -> hi/lo bf16 in LDS ----
#pragma unroll
        for (int i = 0; i < 4; ++i) {
            int f   = tid + i * 256;       // 0..1023 float4 slots
            int row = f >> 3;              // 0..127
            int c4  = (f & 7) << 2;        // 0,4,...,28
            const float4 v = *reinterpret_cast<const float4*>(
                &X[(size_t)(bm + row) * IN_DIM + k0 + c4]);
            float vv[4] = {v.x, v.y, v.z, v.w};
#pragma unroll
            for (int j = 0; j < 4; ++j) {
                __bf16 h = (__bf16)vv[j];
                Ah[row * APITCH + c4 + j] = h;
                Al[row * APITCH + c4 + j] = (__bf16)(vv[j] - (float)h);
            }
        }
        // ---- stage B tile: 32x128 f32 -> transposed hi/lo bf16 in LDS ----
#pragma unroll
        for (int i = 0; i < 4; ++i) {
            int f   = tid + i * 256;
            int row = f >> 5;              // k: 0..31
            int c4  = (f & 31) << 2;       // n: 0,4,...,124
            const float4 v = *reinterpret_cast<const float4*>(
                &W[(size_t)(k0 + row) * OUT_DIM + bn + c4]);
            float vv[4] = {v.x, v.y, v.z, v.w};
#pragma unroll
            for (int j = 0; j < 4; ++j) {
                __bf16 h = (__bf16)vv[j];
                Bh[(c4 + j) * BPITCH + row] = h;
                Bl[(c4 + j) * BPITCH + row] = (__bf16)(vv[j] - (float)h);
            }
        }
        // prefetch next K-tile (global_prefetch_b8 path)
        if (k0 + BK < IN_DIM) {
            __builtin_prefetch(&X[(size_t)(bm + (tid >> 3)) * IN_DIM + k0 + BK], 0, 3);
            __builtin_prefetch(&W[(size_t)(k0 + BK + (tid >> 5)) * OUT_DIM + bn], 0, 3);
        }
        __syncthreads();

        // ---- load fragments (ISA 16-bit A 16x32 / B 32x16 layouts) ----
        v16bf ah[2], al[2], bh[4], bl[4];
#pragma unroll
        for (int mi = 0; mi < 2; ++mi) {
            int m = wm * 32 + mi * 16 + l16;
            const v8bf a0 = *reinterpret_cast<const v8bf*>(&Ah[m * APITCH + 8 * half]);
            const v8bf a1 = *reinterpret_cast<const v8bf*>(&Ah[m * APITCH + 16 + 8 * half]);
            ah[mi] = __builtin_shufflevector(a0, a1, 0,1,2,3,4,5,6,7,8,9,10,11,12,13,14,15);
            const v8bf c0 = *reinterpret_cast<const v8bf*>(&Al[m * APITCH + 8 * half]);
            const v8bf c1 = *reinterpret_cast<const v8bf*>(&Al[m * APITCH + 16 + 8 * half]);
            al[mi] = __builtin_shufflevector(c0, c1, 0,1,2,3,4,5,6,7,8,9,10,11,12,13,14,15);
        }
#pragma unroll
        for (int ni = 0; ni < 4; ++ni) {
            int n = wn * 64 + ni * 16 + l16;
            const v8bf b0 = *reinterpret_cast<const v8bf*>(&Bh[n * BPITCH + 8 * half]);
            const v8bf b1 = *reinterpret_cast<const v8bf*>(&Bh[n * BPITCH + 16 + 8 * half]);
            bh[ni] = __builtin_shufflevector(b0, b1, 0,1,2,3,4,5,6,7,8,9,10,11,12,13,14,15);
            const v8bf d0 = *reinterpret_cast<const v8bf*>(&Bl[n * BPITCH + 8 * half]);
            const v8bf d1 = *reinterpret_cast<const v8bf*>(&Bl[n * BPITCH + 16 + 8 * half]);
            bl[ni] = __builtin_shufflevector(d0, d1, 0,1,2,3,4,5,6,7,8,9,10,11,12,13,14,15);
        }

        // ---- 24 WMMAs: compensated bf16 product ----
#pragma unroll
        for (int mi = 0; mi < 2; ++mi) {
#pragma unroll
            for (int ni = 0; ni < 4; ++ni) {
                acc[mi][ni] = __builtin_amdgcn_wmma_f32_16x16x32_bf16(
                    false, ah[mi], false, bh[ni], (short)0, acc[mi][ni], false, false);
                acc[mi][ni] = __builtin_amdgcn_wmma_f32_16x16x32_bf16(
                    false, ah[mi], false, bl[ni], (short)0, acc[mi][ni], false, false);
                acc[mi][ni] = __builtin_amdgcn_wmma_f32_16x16x32_bf16(
                    false, al[mi], false, bh[ni], (short)0, acc[mi][ni], false, false);
            }
        }
        __syncthreads();
    }

    // ---- epilogue: + bias + hd, tanh, store ----
#pragma unroll
    for (int mi = 0; mi < 2; ++mi) {
#pragma unroll
        for (int ni = 0; ni < 4; ++ni) {
            int n = bn + wn * 64 + ni * 16 + l16;
            float add = bias[n] + hd[n];
#pragma unroll
            for (int r = 0; r < 8; ++r) {
                int m = bm + wm * 32 + mi * 16 + half * 8 + r;  // C/D VGPR layout
                out[(size_t)m * OUT_DIM + n] = tanhf(acc[mi][ni][r] + add);
            }
        }
    }
}

// ---------------------------------------------------------------------------
extern "C" void kernel_launch(void* const* d_in, const int* in_sizes, int n_in,
                              void* d_out, int out_size, void* d_ws, size_t ws_size,
                              hipStream_t stream) {
    (void)in_sizes; (void)n_in; (void)out_size; (void)ws_size;
    const float* x    = (const float*)d_in[0];
    const float* hdw  = (const float*)d_in[1];
    const float* wcls = (const float*)d_in[2];
    const float* bcls = (const float*)d_in[3];
    float* out = (float*)d_out;
    float* hd  = (float*)d_ws;   // OUT_DIM floats of scratch

    hd_probs_kernel<<<OUT_DIM / 256, 256, 0, stream>>>(hdw, hd);

    dim3 grid(OUT_DIM / BN, BATCH / BM);
    fused_gemm_tanh_kernel<<<grid, 256, 0, stream>>>(x, wcls, bcls, hd, out);
}
